// MessagePassing_13666585936093
// MI455X (gfx1250) — matrix operand order repaired and checked by reading, not compiled, not run on hardware
//
#include <hip/hip_runtime.h>

// ---------------------------------------------------------------------------
// GNN message passing on gfx1250 (MI455X), wave32 + v_wmma_f32_16x16x32_bf16
// + Tensor Data Mover for weight staging.
//   inp[E,288] = [n_embed[snd] | n_embed[rcv] | e_embed]   (gathered, fp32)
//   h   = tanh(inp @ W1 + b1)   [E,192]
//   msg = h @ W2 + b2           [E,128]
//   out = segment_mean(msg, senders sorted, N nodes)
// One wave owns 32 edges (two 16-row M tiles sharing every B fragment ->
// 1 ds_load_b128 pair feeds 2 WMMAs).  Block = 8 waves = 256 edges.  Weights
// are pre-swizzled to bf16 WMMA B-fragment layout (prep kernel -> d_ws), then
// DMA'd into LDS with one TENSOR_LOAD_TO_LDS per block (in-kernel conversion
// fallback if d_ws too small).  Sorted senders -> 32-bit run mask via ballot,
// in-register segmented reduce, f32 atomics once per distinct sender per tile.
// ---------------------------------------------------------------------------

typedef __attribute__((ext_vector_type(16))) __bf16 v16bf;
typedef __attribute__((ext_vector_type(8)))  __bf16 v8bf;
typedef __attribute__((ext_vector_type(8)))  float  v8f;
typedef unsigned int u32x4 __attribute__((ext_vector_type(4)));
typedef unsigned int u32x8 __attribute__((ext_vector_type(8)));

#define D_NODE 128
#define D_EDGE 32
#define D_HID  192
#define D_OUT  128
#define KC1    9                 // 288/32 k-chunks for GEMM1
#define NT1    12                // 192/16 n-tiles  for GEMM1
#define KC2    6                 // 192/32 k-chunks for GEMM2
#define NT2    8                 // 128/16 n-tiles  for GEMM2
#define WAVES  8
#define BLOCK  256
#define EDGES_PER_WAVE 32        // two 16-row M tiles
#define EDGES_PER_GROUP (WAVES*EDGES_PER_WAVE)   // 256

#define W1F_ELEMS (KC1*NT1*32*16)     // 55296 bf16
#define W2F_ELEMS (KC2*NT2*32*16)     // 24576 bf16
#define WF_ELEMS  (W1F_ELEMS + W2F_ELEMS)        // 79872
#define HT_ELEMS  (WAVES*EDGES_PER_WAVE*D_HID)   // 49152 bf16
#define SMEM_BYTES ((WF_ELEMS + HT_ELEMS)*2)     // 258048
#define TDM_ELEMS (WF_ELEMS*2/8)                 // 19968 x 8-byte elements

// Native CDNA5 v_tanh_f32 if exposed; else 1 - 2/(exp2(2x*log2e)+1) from
// single-instruction transcendentals (v_exp_f32 / v_rcp_f32).
__device__ __forceinline__ float fast_tanh(float x) {
#if __has_builtin(__builtin_amdgcn_tanhf)
    return __builtin_amdgcn_tanhf(x);
#elif __has_builtin(__builtin_amdgcn_tanh_f32)
    return __builtin_amdgcn_tanh_f32(x);
#else
#if __has_builtin(__builtin_amdgcn_exp2f) && __has_builtin(__builtin_amdgcn_rcpf)
    const float e = __builtin_amdgcn_exp2f(2.885390081777927f * x);  // e^(2x)
    return 1.0f - 2.0f * __builtin_amdgcn_rcpf(e + 1.0f);
#else
    const float e = __expf(2.0f * x);
    return 1.0f - 2.0f / (e + 1.0f);
#endif
#endif
}

// B-frag (16x16x32 bf16): lane L element h holds B[K = 16*(L/16)+h][N = L%16].
// Map flat (row,col) of a [rows x cols] weight into fragment-array index.
__device__ __forceinline__ int frag_index(int row, int col, int ntiles) {
    const int kb = row >> 5, kr = row & 31;
    const int lane = ((kr >> 4) << 4) | (col & 15);
    const int h = kr & 15, nt = col >> 4;
    return (((kb*ntiles + nt)*32 + lane) << 4) + h;
}

// one-time: fp32 weights -> bf16 fragment-swizzled arrays in workspace
__global__ void prep_kernel(const float* __restrict__ W1,
                            const float* __restrict__ W2,
                            __bf16* __restrict__ wsFrag) {
    const int idx = blockIdx.x*blockDim.x + threadIdx.x;
    if (idx < W1F_ELEMS) {
        const int row = idx / D_HID, col = idx % D_HID;
        wsFrag[frag_index(row, col, NT1)] = (__bf16)W1[idx];
    } else if (idx < WF_ELEMS) {
        const int i2 = idx - W1F_ELEMS;
        const int row = i2 / D_OUT, col = i2 % D_OUT;
        wsFrag[W1F_ELEMS + frag_index(row, col, NT2)] = (__bf16)W2[i2];
    }
}

__global__ void __launch_bounds__(BLOCK)
mp_kernel(const float* __restrict__ nEmb, const float* __restrict__ eEmb,
          const int*   __restrict__ senders, const int* __restrict__ receivers,
          const float* __restrict__ W1, const float* __restrict__ b1,
          const float* __restrict__ W2, const float* __restrict__ b2,
          const __bf16* __restrict__ wsFrag,   // nullptr -> convert in kernel
          float* __restrict__ out, int nEdges)
{
    extern __shared__ unsigned char smem[];
    __bf16* w1f = (__bf16*)smem;              // B-fragment swizzled W1 (bf16)
    __bf16* w2f = w1f + W1F_ELEMS;            // B-fragment swizzled W2 (bf16)
    __bf16* hT  = w2f + W2F_ELEMS;            // per-wave 32x192 hidden tiles

    const int tid = threadIdx.x;

    if (wsFrag) {
        // ---- stage 156 KB of fragment-swizzled weights via Tensor Data Mover.
        // 1-D descriptor: 19968 x 8B elements, global(ws) -> LDS.  Issued by
        // wave 0 only (TDM ignores EXEC); tracked with TENSORcnt.
        if (tid < 32) {
            const unsigned lds0 = (unsigned)(size_t)(void*)smem; // LDS aperture: offset in addr[31:0]
            const unsigned long long ga = (unsigned long long)(size_t)wsFrag;
            u32x4 g0;
            g0[0] = 1u;                                        // count=1, load, no gather
            g0[1] = lds0;                                      // lds_addr
            g0[2] = (unsigned)(ga & 0xFFFFFFFFu);              // global_addr[31:0]
            g0[3] = (unsigned)((ga >> 32) & 0x01FFFFFFu)       // global_addr[56:32]
                  | (2u << 30);                                // type = 2 ("image")
            u32x8 g1;
            g1[0] = 3u << 16;                                  // data_size = 8B, no masks
            g1[1] = ((unsigned)TDM_ELEMS & 0xFFFFu) << 16;     // tensor_dim0[15:0]
            g1[2] = (((unsigned)TDM_ELEMS >> 16) & 0xFFFFu)    // tensor_dim0[31:16]
                  | (1u << 16);                                // tensor_dim1 = 1
            g1[3] = ((unsigned)TDM_ELEMS & 0xFFFFu) << 16;     // tile_dim0 = whole row
            g1[4] = 1u;                                        // tile_dim1 = 1
            g1[5] = (unsigned)TDM_ELEMS;                       // tensor_dim0_stride
            g1[6] = 0u;
            g1[7] = 0u;
            asm volatile("tensor_load_to_lds %0, %1" :: "s"(g0), "s"(g1) : "memory");
            __builtin_amdgcn_s_wait_tensorcnt(0);
        }
    } else {                                  // fallback: convert + swizzle
        for (int idx = tid; idx < W1F_ELEMS; idx += BLOCK) {
            const int row = idx / D_HID, col = idx % D_HID;
            w1f[frag_index(row, col, NT1)] = (__bf16)W1[idx];
        }
        for (int idx = tid; idx < W2F_ELEMS; idx += BLOCK) {
            const int row = idx / D_OUT, col = idx % D_OUT;
            w2f[frag_index(row, col, NT2)] = (__bf16)W2[idx];
        }
    }
    __syncthreads();

    const int lane = tid & 31, wid = tid >> 5;
    const int half = lane >> 4;          // lane group (0: lanes 0-15, 1: 16-31)
    const int l16  = lane & 15;          // = A-row M for A-frags, = col N for C/D
    const int koff = half * 16;
    __bf16* hWave = hT + wid*(EDGES_PER_WAVE*D_HID);

    // ---- group-invariant per-lane biases (C/D column N = lane%16)
    float bias1v[NT1], bias2v[NT2];
#pragma unroll
    for (int nt = 0; nt < NT1; ++nt) bias1v[nt] = b1[nt*16 + l16];
#pragma unroll
    for (int nt = 0; nt < NT2; ++nt) bias2v[nt] = b2[nt*16 + l16];

    const int ngroups = nEdges / EDGES_PER_GROUP;   // exact: 320000/256 = 1250
    for (int grp = blockIdx.x; grp < ngroups; grp += gridDim.x) {
        const int edgeBase = grp*EDGES_PER_GROUP + wid*EDGES_PER_WAVE;
        const int e0 = edgeBase + l16;          // M tile 0 row
        const int e1 = e0 + 16;                 // M tile 1 row
        const int s0 = senders[e0],   s1 = senders[e1];
        const int r0 = receivers[e0], r1 = receivers[e1];

        // lane L owns sorted row L's sender; 32-bit run-start mask via ballot
        const int rowSend = (lane < 16) ? s0 : s1;
        const int sprev = __shfl_up(rowSend, 1, 32);
        const bool isStart = (lane == 0) || (sprev != rowSend);
        const unsigned runMask = (unsigned)__ballot(isStart);

        // ---- build A fragments (two 16x288 tiles), bf16
        // A-frag: lane L element h holds A[M=L%16][K = 16*(L/16)+h]
        v16bf a1[2][KC1];
#pragma unroll
        for (int t = 0; t < 2; ++t) {
            const int s = t ? s1 : s0, r = t ? r1 : r0, e = t ? e1 : e0;
#pragma unroll
            for (int c = 0; c < KC1; ++c) {
                const float* base;
                if (c < 4)      base = nEmb + (size_t)s*D_NODE + c*32;
                else if (c < 8) base = nEmb + (size_t)r*D_NODE + (c-4)*32;
                else            base = eEmb + (size_t)e*D_EDGE;
                const float* p0 = base + koff;  // two contiguous 8-float runs
#pragma unroll
                for (int i = 0; i < 8; ++i) {
                    a1[t][c][i]     = (__bf16)p0[i];
                    a1[t][c][i + 8] = (__bf16)p0[8 + i];
                }
            }
        }

        // ---- GEMM1: h = tanh(inp @ W1 + b1) -> bf16 tiles in LDS
        // 2 n-tiles x 2 M-tiles: 4 independent chains, each B frag feeds 2 WMMAs
        for (int ntp = 0; ntp < NT1; ntp += 2) {
            v8f acA0, acA1, acB0, acB1;
#pragma unroll
            for (int i = 0; i < 8; ++i) {
                acA0[i] = bias1v[ntp];     acA1[i] = bias1v[ntp];
                acB0[i] = bias1v[ntp + 1]; acB1[i] = bias1v[ntp + 1];
            }
#pragma unroll
            for (int k = 0; k < KC1; ++k) {
                const v16bf bA = *(const v16bf*)(w1f + (((k*NT1 + ntp    )*32 + lane) << 4));
                const v16bf bB = *(const v16bf*)(w1f + (((k*NT1 + ntp + 1)*32 + lane) << 4));
                acA0 = __builtin_amdgcn_wmma_f32_16x16x32_bf16(false, a1[0][k], false, bA, (short)0, acA0, false, false);
                acA1 = __builtin_amdgcn_wmma_f32_16x16x32_bf16(false, a1[1][k], false, bA, (short)0, acA1, false, false);
                acB0 = __builtin_amdgcn_wmma_f32_16x16x32_bf16(false, a1[0][k], false, bB, (short)0, acB0, false, false);
                acB1 = __builtin_amdgcn_wmma_f32_16x16x32_bf16(false, a1[1][k], false, bB, (short)0, acB1, false, false);
            }
#pragma unroll
            for (int v = 0; v < 8; ++v) {       // C/D: M = 16*t + v + 8*half
                __bf16* hr0 = hWave + (v + 8*half)*D_HID + l16;
                __bf16* hr1 = hr0 + 16*D_HID;
                hr0[ntp*16]       = (__bf16)fast_tanh(acA0[v]);
                hr1[ntp*16]       = (__bf16)fast_tanh(acA1[v]);
                hr0[(ntp + 1)*16] = (__bf16)fast_tanh(acB0[v]);
                hr1[(ntp + 1)*16] = (__bf16)fast_tanh(acB1[v]);
            }
        }

        // ---- reload h as A fragments (same per-lane two-run pattern, from LDS)
        v16bf a2[2][KC2];
#pragma unroll
        for (int t = 0; t < 2; ++t) {
            const __bf16* hRow = hWave + (16*t + l16)*D_HID;
#pragma unroll
            for (int kc = 0; kc < KC2; ++kc) {
                const v8bf lo = *(const v8bf*)(hRow + kc*32 + koff);
                const v8bf hi = *(const v8bf*)(hRow + kc*32 + 8 + koff);
#pragma unroll
                for (int i = 0; i < 8; ++i) { a2[t][kc][i] = lo[i]; a2[t][kc][i+8] = hi[i]; }
            }
        }

        // ---- GEMM2 (+bias in C init), segmented reduce over sorted-sender runs
        for (int ntp = 0; ntp < NT2; ntp += 2) {
            v8f acA0, acA1, acB0, acB1;
#pragma unroll
            for (int i = 0; i < 8; ++i) {
                acA0[i] = bias2v[ntp];     acA1[i] = bias2v[ntp];
                acB0[i] = bias2v[ntp + 1]; acB1[i] = bias2v[ntp + 1];
            }
#pragma unroll
            for (int kc = 0; kc < KC2; ++kc) {
                const v16bf bA = *(const v16bf*)(w2f + (((kc*NT2 + ntp    )*32 + lane) << 4));
                const v16bf bB = *(const v16bf*)(w2f + (((kc*NT2 + ntp + 1)*32 + lane) << 4));
                acA0 = __builtin_amdgcn_wmma_f32_16x16x32_bf16(false, a2[0][kc], false, bA, (short)0, acA0, false, false);
                acA1 = __builtin_amdgcn_wmma_f32_16x16x32_bf16(false, a2[1][kc], false, bA, (short)0, acA1, false, false);
                acB0 = __builtin_amdgcn_wmma_f32_16x16x32_bf16(false, a2[0][kc], false, bB, (short)0, acB0, false, false);
                acB1 = __builtin_amdgcn_wmma_f32_16x16x32_bf16(false, a2[1][kc], false, bB, (short)0, acB1, false, false);
            }
            unsigned m = runMask;
            while (m) {                         // uniform loop over sender runs
                const int row = __ffs(m) - 1;
                m &= m - 1;
                const int end = m ? (__ffs(m) - 1) : 32;
                const int g = __shfl(rowSend, row, 32);   // sender of this run
                float sA = 0.f, sB = 0.f;
#pragma unroll
                for (int t = 0; t < 2; ++t)
#pragma unroll
                    for (int v = 0; v < 8; ++v) {
                        const int mm = 16*t + v + 8*half; // global row owned here
                        if (mm >= row && mm < end) {
                            sA += t ? acA1[v] : acA0[v];
                            sB += t ? acB1[v] : acB0[v];
                        }
                    }
                sA += __shfl_xor(sA, 16, 32);   // combine lane halves (M, M+8)
                sB += __shfl_xor(sB, 16, 32);
                if (lane < 16) {
                    float* op = out + (size_t)g*D_OUT + ntp*16 + lane;
                    atomicAdd(op, sA);
                    atomicAdd(op + 16, sB);
                }
            }
        }
    }
}

__global__ void zero_kernel(float* __restrict__ p, int n) {
    const int i = blockIdx.x*blockDim.x + threadIdx.x;
    if (i < n) p[i] = 0.f;
}

// counts via binary search in sorted senders (no atomics -> deterministic)
__global__ void finalize_kernel(const int* __restrict__ senders,
                                float* __restrict__ out, int nEdges) {
    const int node = blockIdx.x;
    int lo = 0, hi = nEdges;
    while (lo < hi) { int m = (lo + hi) >> 1; if (senders[m] <  node) lo = m + 1; else hi = m; }
    int lo2 = lo, hi2 = nEdges;
    while (lo2 < hi2) { int m = (lo2 + hi2) >> 1; if (senders[m] <= node) lo2 = m + 1; else hi2 = m; }
    const float inv = 1.0f / fmaxf((float)(lo2 - lo), 1.0f);
    out[(size_t)node*D_OUT + threadIdx.x] *= inv;
}

extern "C" void kernel_launch(void* const* d_in, const int* in_sizes, int n_in,
                              void* d_out, int out_size, void* d_ws, size_t ws_size,
                              hipStream_t stream) {
    const float* nEmb      = (const float*)d_in[0];
    const float* eEmb      = (const float*)d_in[1];
    const int*   senders   = (const int*)  d_in[2];
    const int*   receivers = (const int*)  d_in[3];
    const float* W1        = (const float*)d_in[4];
    const float* b1        = (const float*)d_in[5];
    const float* W2        = (const float*)d_in[6];
    const float* b2        = (const float*)d_in[7];
    float* out = (float*)d_out;

    const int nEdges = in_sizes[2];
    const int nNodes = in_sizes[0] / D_NODE;

    zero_kernel<<<(out_size + 255)/256, 256, 0, stream>>>(out, out_size);

    __bf16* wsFrag = nullptr;
    if (ws_size >= (size_t)WF_ELEMS * sizeof(__bf16)) {     // 156 KB scratch
        wsFrag = (__bf16*)d_ws;
        prep_kernel<<<(WF_ELEMS + 255)/256, 256, 0, stream>>>(W1, W2, wsFrag);
    }

    const int ngroups = nEdges / EDGES_PER_GROUP;
    int grid = ngroups < 320 ? ngroups : 320;   // grid-stride; 1 block per WGP (252 KB LDS)
    mp_kernel<<<grid, BLOCK, SMEM_BYTES, stream>>>(nEmb, eEmb, senders, receivers,
                                                   W1, b1, W2, b2, wsFrag, out, nEdges);

    finalize_kernel<<<nNodes, D_OUT, 0, stream>>>(senders, out, nEdges);
}
